// VoxelRasterizer_26422638805067
// MI455X (gfx1250) — matrix operand order repaired and checked by reading, not compiled, not run on hardware
//
#include <hip/hip_runtime.h>
#include <hip/hip_bf16.h>

#define NVOX   1024
#define NRAYS  2048
#define NSH    9
#define KPAD   16
#define NCOL   (NVOX * 3)      // 3072 voxel-channel columns
#define SSAMP  8
#define TILES_M (NRAYS / 16)   // 128
#define TILES_N (NCOL  / 16)   // 192

typedef _Float16 half16 __attribute__((ext_vector_type(16)));
typedef float    float8 __attribute__((ext_vector_type(8)));
typedef float    float4v __attribute__((ext_vector_type(4)));

// ---------------- SH basis per ray, padded K=16 ----------------
__global__ void vr_sh_kernel(const float* __restrict__ dirs, float* __restrict__ shPad) {
    int r = blockIdx.x * blockDim.x + threadIdx.x;
    if (r >= NRAYS) return;
    float x = dirs[r * 3 + 0], y = dirs[r * 3 + 1], z = dirs[r * 3 + 2];
    float s[KPAD];
    s[0] = 1.f; s[1] = y; s[2] = z; s[3] = x;
    s[4] = x * y; s[5] = y * z; s[6] = 3.f * z * z - 1.f; s[7] = x * z; s[8] = x * x - y * y;
    #pragma unroll
    for (int k = NSH; k < KPAD; ++k) s[k] = 0.f;
    #pragma unroll
    for (int k = 0; k < KPAD; ++k) shPad[r * KPAD + k] = s[k];
}

// ---------------- coeffs -> f16, column-major padded K=16 ----------------
__global__ void vr_col_kernel(const float* __restrict__ colors, _Float16* __restrict__ colH) {
    int j = blockIdx.x * blockDim.x + threadIdx.x;
    if (j >= NCOL) return;
    int vox = j / 3, ch = j % 3;
    #pragma unroll
    for (int k = 0; k < KPAD; ++k) {
        float v = (k < NSH) ? colors[vox * 3 * NSH + ch * NSH + k] : 0.f;
        colH[j * KPAD + k] = (_Float16)v;
    }
}

// ---------------- per (ray,voxel): AABB + closed-form 8-sample composite ----------------
__global__ void vr_pairs_kernel(const float* __restrict__ pos, const float* __restrict__ sizes,
                                const float* __restrict__ dens,
                                const float* __restrict__ rayo, const float* __restrict__ rayd,
                                float* __restrict__ keyA, float* __restrict__ wsA,
                                float* __restrict__ depA) {
    int p = blockIdx.x * blockDim.x + threadIdx.x;
    if (p >= NRAYS * NVOX) return;
    int ray = p >> 10;
    int vox = p & (NVOX - 1);
    float hs = 0.5f * sizes[vox];
    float tn = -3.0e38f, tf = 3.0e38f;
    #pragma unroll
    for (int c = 0; c < 3; ++c) {
        float o = rayo[ray * 3 + c], d = rayd[ray * 3 + c];
        float inv = 1.f / d;
        float pc = pos[vox * 3 + c];
        float t0 = (pc - hs - o) * inv;
        float t1 = (pc + hs - o) * inv;
        tn = fmaxf(tn, fminf(t0, t1));
        tf = fminf(tf, fmaxf(t0, t1));
    }
    bool valid = (tf > tn) && (tf > 0.f);
    float tns = valid ? tn : 0.f;
    float tfs = valid ? tf : 1.f;
    float sigma = __expf(dens[vox]);
    float delta = (tfs - tns) * (1.f / SSAMP);
    float alpha = 1.f - __expf(-sigma * delta);
    float base = 1.f - alpha + 1e-8f;
    float pw = 1.f, ws = 0.f, dep = 0.f;
    #pragma unroll
    for (int i = 0; i < SSAMP; ++i) {
        float w  = alpha * pw;
        float ts = tns + (tfs - tns) * ((float)i * (1.f / (SSAMP - 1)));
        ws  += w;
        dep += w * ts;
        pw  *= base;
    }
    keyA[p] = valid ? tn : __builtin_inff();
    wsA[p]  = valid ? ws : 0.f;
    depA[p] = valid ? dep : 0.f;
}

// ---------------- WMMA GEMM: rgb_vox = sigmoid(sh @ coeffs), K padded 9->32 ----------------
__global__ void vr_gemm_kernel(const float* __restrict__ shPad, const _Float16* __restrict__ colH,
                               float* __restrict__ rgbV) {
    int wid  = blockIdx.x * (blockDim.x >> 5) + (threadIdx.x >> 5);
    int lane = threadIdx.x & 31;
    if (wid >= TILES_M * TILES_N) return;
    int tm = wid / TILES_N;
    int tn = wid % TILES_N;
    int m  = lane & 15;
    bool hiHalf = lane >= 16;

    // A 16x32 f16: lanes 0-15 hold K=0-15 (elem e = K e);
    //              lanes 16-31 hold K=16-31 (all zero). K>=9 is zero.
    int rayA  = tm * 16 + m;
    int kbase = 0;
    const float4v* ap = (const float4v*)(shPad + rayA * KPAD + kbase);
    float4v a0 = ap[0];
    float4v a1 = ap[1];
    float4v a2 = ap[2];
    float4v a3 = ap[3];
    half16 a;
    #pragma unroll
    for (int e = 0; e < 4; ++e) a[e]     = hiHalf ? (_Float16)0.f : (_Float16)a0[e];
    #pragma unroll
    for (int e = 0; e < 4; ++e) a[4 + e] = hiHalf ? (_Float16)0.f : (_Float16)a1[e];
    #pragma unroll
    for (int e = 0; e < 4; ++e) a[8 + e] = hiHalf ? (_Float16)0.f : (_Float16)a2[e];
    #pragma unroll
    for (int e = 0; e < 4; ++e) a[12 + e] = hiHalf ? (_Float16)0.f : (_Float16)a3[e];

    // B 32x16 f16: lanes 0-15 hold column N=lane, K=0-15; lanes 16-31 hold K=16-31 (all zero).
    int col = tn * 16 + m;
    half16 b = {};
    if (!hiHalf) {
        const half16* bp = (const half16*)(colH + col * KPAD);
        b = *bp;
    }

    float8 c = {};
    c = __builtin_amdgcn_wmma_f32_16x16x32_f16(false, a, false, b, (short)0, c, false, false);

    // C/D layout: VGPR r, lanes 0-15 -> M=r, N=lane; lanes 16-31 -> M=r+8, N=lane-16.
    #pragma unroll
    for (int r = 0; r < 8; ++r) {
        int row = r + (hiHalf ? 8 : 0);
        int ray = tm * 16 + row;
        // fast sigmoid: v_rcp_f32 instead of the IEEE div_scale/div_fixup sequence
        float v = __builtin_amdgcn_rcpf(1.f + __expf(-c[r]));
        rgbV[ray * NCOL + tn * 16 + m] = v;
    }
}

// ---------------- per-ray: LDS bitonic argsort + scan-composite + reduce ----------------
__global__ void vr_composite_kernel(const float* __restrict__ keyA, const float* __restrict__ wsA,
                                    const float* __restrict__ depA, const float* __restrict__ rgbV,
                                    float* __restrict__ out) {
    __shared__ unsigned long long skey[NVOX];
    __shared__ float sscan[256];
    int ray = blockIdx.x;
    int tid = threadIdx.x;

    // load keys, encode to order-preserving u32, pack index in low bits (stable sort)
    for (int i = tid; i < NVOX; i += 256) {
        unsigned int bb = __float_as_uint(keyA[ray * NVOX + i]);
        unsigned int u  = bb ^ ((bb >> 31) ? 0xFFFFFFFFu : 0x80000000u);
        skey[i] = ((unsigned long long)u << 32) | (unsigned int)i;
    }
    __syncthreads();

    // bitonic sort ascending (invalid -> +inf at the end)
    for (int k = 2; k <= NVOX; k <<= 1) {
        for (int j = k >> 1; j > 0; j >>= 1) {
            for (int i = tid; i < NVOX; i += 256) {
                int ixj = i ^ j;
                if (ixj > i) {
                    unsigned long long av = skey[i], bv = skey[ixj];
                    bool asc = ((i & k) == 0);
                    if ((av > bv) == asc) { skey[i] = bv; skey[ixj] = av; }
                }
            }
            __syncthreads();
        }
    }

    // each thread owns 4 contiguous sorted positions
    int q0 = tid * 4;
    int idx[4]; float wsv[4], depv[4];
    float prod = 1.f;
    #pragma unroll
    for (int e = 0; e < 4; ++e) {
        int id  = (int)(unsigned int)(skey[q0 + e] & 0xFFFFFFFFull);
        idx[e]  = id;
        wsv[e]  = wsA[ray * NVOX + id];
        depv[e] = depA[ray * NVOX + id];
        prod   *= (1.f - wsv[e]);
    }

    // block exclusive prefix product of (1 - ws) -> T_before
    sscan[tid] = prod;
    __syncthreads();
    for (int off = 1; off < 256; off <<= 1) {
        float v = sscan[tid];
        float m = (tid >= off) ? sscan[tid - off] : 1.f;
        __syncthreads();
        sscan[tid] = v * m;
        __syncthreads();
    }
    float T = (tid > 0) ? sscan[tid - 1] : 1.f;
    __syncthreads();

    float racc = 0.f, gacc = 0.f, bacc = 0.f, dacc = 0.f, wacc = 0.f;
    #pragma unroll
    for (int e = 0; e < 4; ++e) {
        if (T >= 0.01f) {                     // early-stop mask (proc)
            float contrib = T * wsv[e];
            const float* rp = rgbV + ray * NCOL + idx[e] * 3;
            racc += contrib * rp[0];
            gacc += contrib * rp[1];
            bacc += contrib * rp[2];
            dacc += T * depv[e];
            wacc += contrib;
        }
        T *= (1.f - wsv[e]);                  // cumprod advances regardless of proc
    }

    // 5 tree reductions (r, g, b, depth, weight)
    float vals[5] = {racc, gacc, bacc, dacc, wacc};
    float res[5];
    #pragma unroll
    for (int s = 0; s < 5; ++s) {
        sscan[tid] = vals[s];
        __syncthreads();
        for (int off = 128; off > 0; off >>= 1) {
            if (tid < off) sscan[tid] += sscan[tid + off];
            __syncthreads();
        }
        res[s] = sscan[0];
        __syncthreads();
    }

    if (tid == 0) {
        // encoded(+inf) = 0xFF800000; any smaller min-key means a hit
        bool hasHit = ((unsigned int)(skey[0] >> 32)) < 0xFF800000u;
        out[ray * 3 + 0] = res[0];
        out[ray * 3 + 1] = res[1];
        out[ray * 3 + 2] = res[2];
        out[NRAYS * 3 + ray] = hasHit ? res[3] : 100.0f;   // depth (FAR_PLANE fallback)
        out[NRAYS * 4 + ray] = res[4];                     // weight_acc
    }
}

extern "C" void kernel_launch(void* const* d_in, const int* in_sizes, int n_in,
                              void* d_out, int out_size, void* d_ws, size_t ws_size,
                              hipStream_t stream) {
    (void)in_sizes; (void)n_in; (void)out_size; (void)ws_size;
    const float* pos    = (const float*)d_in[0];
    const float* sizes  = (const float*)d_in[1];
    const float* dens   = (const float*)d_in[2];
    const float* colors = (const float*)d_in[3];
    const float* rayo   = (const float*)d_in[4];
    const float* rayd   = (const float*)d_in[5];
    float* out = (float*)d_out;

    char* ws = (char*)d_ws;
    size_t off = 0;
    float*    shPad = (float*)(ws + off);      off += (size_t)NRAYS * KPAD * sizeof(float);     // 128 KB
    _Float16* colH  = (_Float16*)(ws + off);   off += (size_t)NCOL * KPAD * sizeof(_Float16);   //  96 KB
    float*    keyA  = (float*)(ws + off);      off += (size_t)NRAYS * NVOX * sizeof(float);     //   8 MB
    float*    wsA   = (float*)(ws + off);      off += (size_t)NRAYS * NVOX * sizeof(float);     //   8 MB
    float*    depA  = (float*)(ws + off);      off += (size_t)NRAYS * NVOX * sizeof(float);     //   8 MB
    float*    rgbV  = (float*)(ws + off);      off += (size_t)NRAYS * NCOL * sizeof(float);     //  24 MB

    vr_sh_kernel  <<<(NRAYS + 255) / 256, 256, 0, stream>>>(rayd, shPad);
    vr_col_kernel <<<(NCOL  + 255) / 256, 256, 0, stream>>>(colors, colH);
    vr_pairs_kernel<<<(NRAYS * NVOX) / 256, 256, 0, stream>>>(pos, sizes, dens, rayo, rayd,
                                                              keyA, wsA, depA);
    vr_gemm_kernel<<<(TILES_M * TILES_N) / 8, 256, 0, stream>>>(shPad, colH, rgbV);
    vr_composite_kernel<<<NRAYS, 256, 0, stream>>>(keyA, wsA, depA, rgbV, out);
}
